// NonLocalMeans_16509854286240
// MI455X (gfx1250) — compile-verified, matrix operand order checked by reading, not run on hardware
//
#include <hip/hip_runtime.h>
#include <hip/hip_bf16.h>

// ---------------------------------------------------------------------------
// NLM for 1x384x384x3 f32, R=9 (361 offsets), 9x9 patch box, 7x7x3x3 feature
// conv. Kernel A: feature conv as im2col GEMM on WMMA (f16 in, f32 acc).
// Kernel B: tile-resident NLM; async global->LDS fills, float4-padded pixels
// for ds_load_b128, separable box filter, v_exp per weight.
// ---------------------------------------------------------------------------

#define HW  384
#define CH  3
#define PR  4            // patch radius (PF=9)
#define RR  9            // search radius
#define TT  32           // output tile
#define XRG (TT + 2*RR)          // 50 : x region
#define FRG (TT + 2*(RR+PR))     // 58 : feat region
#define DRG (TT + 2*PR)          // 40 : d2 region

typedef __attribute__((ext_vector_type(16))) _Float16 v16h;
typedef __attribute__((ext_vector_type(8)))  float    v8f;
typedef __attribute__((ext_vector_type(4)))  float    v4f;

__device__ __forceinline__ int clampi(int v, int lo, int hi) {
  return v < lo ? lo : (v > hi ? hi : v);
}

// ---- async global->LDS (gfx1250), with safe fallback --------------------
#if defined(__gfx1250__) && __has_builtin(__builtin_amdgcn_global_load_async_to_lds_b32)
#define NLM_ASYNC_LDS 1
#endif

__device__ __forceinline__ void lds_fill_f32(float* lds, const float* g) {
#if defined(NLM_ASYNC_LDS)
  __builtin_amdgcn_global_load_async_to_lds_b32(
      (__attribute__((address_space(1))) int*)g,
      (__attribute__((address_space(3))) int*)lds, 0, 0);
#else
  *lds = *g;
#endif
}

__device__ __forceinline__ void lds_fill_wait() {
#if defined(NLM_ASYNC_LDS)
#if __has_builtin(__builtin_amdgcn_s_wait_asynccnt)
  __builtin_amdgcn_s_wait_asynccnt(0);
#else
  asm volatile("s_wait_asynccnt 0" ::: "memory");
#endif
#endif
  __syncthreads();
}

// ---------------------------------------------------------------------------
// Kernel A: feat = conv7x7_edge(x, ff) as WMMA GEMM.
// M = pixel (row-major), K = (i*7+j)*3+ci (147, zero-padded to 160), N = oc.
// One wave handles a 16-pixel block; 8 waves / block. K loop fully unrolled:
// 5 x v_wmma_f32_16x16x32_f16 with gathers overlapped across steps.
// ---------------------------------------------------------------------------
__global__ __launch_bounds__(256) void feat_conv_wmma(
    const float* __restrict__ x, const float* __restrict__ ff,
    float* __restrict__ feat) {
  const int lane = threadIdx.x & 31;
  const int wave = threadIdx.x >> 5;
  const int pix0 = (blockIdx.x * 8 + wave) * 16;
  const int l16  = lane & 15;
  const bool hi  = lane >= 16;

  const int p  = pix0 + l16;           // A-matrix row (pixel) for this lane
  const int py = p / HW, px = p % HW;

  v8f acc = {};
  #pragma unroll
  for (int k0 = 0; k0 < 160; k0 += 32) {
    v16h a, b;
    // 16-bit A 16x32 layout (ISA 7.12.2): lanes0-15 VGPR0-3 = K0..7,
    // VGPR4-7 = K16..23; lanes16-31 shifted by +8 in each group.
    #pragma unroll
    for (int e = 0; e < 16; ++e) {
      const int v = e >> 1, pos = e & 1;
      const int k = k0 + (v < 4 ? 2*v : 16 + 2*(v - 4)) + pos + (hi ? 8 : 0);
      float av = 0.f, bv = 0.f;
      if (k < 147) {
        const int i = k / 21, r = k - i*21, j = r / 3, ci = r - j*3;
        const int yy = clampi(py + i - 3, 0, HW - 1);
        const int xx = clampi(px + j - 3, 0, HW - 1);
        av = x[(yy*HW + xx)*CH + ci];
        if (l16 < CH) bv = ff[k*CH + l16];   // B[K=k][N=l16]
      }
      a[e] = (_Float16)av;
      b[e] = (_Float16)bv;
    }
    acc = __builtin_amdgcn_wmma_f32_16x16x32_f16(
        false, a, false, b, (short)0, acc, false, false);
  }

  // D layout: lane<16 -> VGPR r = D[M=r][N=lane]; lane>=16 -> M=8+r, N=lane-16.
  const int n  = l16;
  const int mb = hi ? 8 : 0;
  if (n < CH) {
    #pragma unroll
    for (int r = 0; r < 8; ++r)
      feat[(pix0 + mb + r)*CH + n] = acc[r];
  }
}

// ---------------------------------------------------------------------------
// Kernel B: tile-resident NLM. 32x32 output tile / 256 threads.
// Pixels stored in LDS as float4 (ch 0..2 + pad) => b128 LDS accesses in the
// hot loop. Per offset: d2 (40x40) -> weighted row sums (40x32) -> col sums +
// exp + accumulate; 2 barriers per offset.
// ---------------------------------------------------------------------------
__global__ __launch_bounds__(256) void nlm_tile_kernel(
    const float* __restrict__ x, const float* __restrict__ feat,
    const float* __restrict__ pfilt, const float* __restrict__ inv_sigma,
    float* __restrict__ out) {
  __shared__ v4f   sx4[XRG*XRG];          // x  region, base tile-9   (40 KB)
  __shared__ v4f   sf4[FRG*FRG];          // ft region, base tile-13  (53.8 KB)
  __shared__ float sd2[DRG*(DRG+1)];      // d2, padded stride        (6.6 KB)
  __shared__ float srs[DRG*(TT+1)];       // row sums, padded stride  (5.3 KB)

  const int tid = threadIdx.x;
  const int ty0 = blockIdx.y * TT, tx0 = blockIdx.x * TT;

  // Separable patch weights (exact for the all-ones 9x9 filter).
  float rw[9], cw[9];
  {
    const float ctr = pfilt[4*9 + 4];
    #pragma unroll
    for (int i = 0; i < 9; ++i) {
      rw[i] = pfilt[4*9 + i];
      cw[i] = pfilt[i*9 + 4] / ctr;
    }
  }

  // -------- LDS fills (clamped, channel-padded to 4), async --------
  {
    float* sxf = (float*)sx4;
    for (int idx = tid; idx < XRG*XRG*4; idx += 256) {
      const int c = idx & 3, t = idx >> 2;          // pad lane c==3 -> dup c0
      const int cu = (c == 3) ? 0 : c;
      const int lx = t % XRG, ly = t / XRG;
      const int gy = clampi(ty0 - RR + ly, 0, HW-1);
      const int gx = clampi(tx0 - RR + lx, 0, HW-1);
      lds_fill_f32(&sxf[idx], &x[(gy*HW + gx)*CH + cu]);
    }
    float* sff = (float*)sf4;
    for (int idx = tid; idx < FRG*FRG*4; idx += 256) {
      const int c = idx & 3, t = idx >> 2;
      const int cu = (c == 3) ? 0 : c;
      const int lx = t % FRG, ly = t / FRG;
      const int gy = clampi(ty0 - (RR+PR) + ly, 0, HW-1);
      const int gx = clampi(tx0 - (RR+PR) + lx, 0, HW-1);
      lds_fill_f32(&sff[idx], &feat[(gy*HW + gx)*CH + cu]);
    }
  }
  lds_fill_wait();

  // -------- hoist d2-point clamped coordinates out of the offset loop ------
  int d2st[7], d2f[7];
  #pragma unroll
  for (int k = 0; k < 7; ++k) {
    const int idx = tid + k*256;
    if (idx < DRG*DRG) {
      const int ly = idx / DRG, lx = idx % DRG;
      const int gy = clampi(ty0 - PR + ly, 0, HW-1);
      const int gx = clampi(tx0 - PR + lx, 0, HW-1);
      d2st[k] = ly*(DRG+1) + lx;
      d2f[k]  = (gy - ty0 + (RR+PR))*FRG + (gx - tx0 + (RR+PR));  // pixel idx
    }
  }

  float num0[4] = {0,0,0,0}, num1[4] = {0,0,0,0}, num2[4] = {0,0,0,0};
  float den[4]  = {0,0,0,0};
  const float isg = inv_sigma[0];
  const int opy = tid >> 5, opx = tid & 31;   // 4 output pixels per thread

  for (int dy = -RR; dy <= RR; ++dy) {
    for (int dx = -RR; dx <= RR; ++dx) {
      const int soff = dy*FRG + dx;            // pixel-index shift in sf4

      // phase 1: d2 over tile +/- 4 (2x ds_load_b128 + 1 store per point)
      #pragma unroll
      for (int k = 0; k < 7; ++k) {
        if (tid + k*256 < DRG*DRG) {
          const v4f c0 = sf4[d2f[k]];
          const v4f c1 = sf4[d2f[k] + soff];
          const float a0 = c0.x - c1.x, a1 = c0.y - c1.y, a2 = c0.z - c1.z;
          sd2[d2st[k]] = a0*a0 + a1*a1 + a2*a2;
        }
      }
      __syncthreads();

      // phase 2: horizontal weighted 9-sum (1280 points, exactly 5/thread)
      #pragma unroll
      for (int k = 0; k < 5; ++k) {
        const int idx = tid + k*256;
        const int ry = idx / TT, rx = idx % TT;
        const float* r = &sd2[ry*(DRG+1) + rx];
        float s = 0.f;
        #pragma unroll
        for (int u = 0; u < 9; ++u) s += rw[u] * r[u];
        srs[ry*(TT+1) + rx] = s;
      }
      __syncthreads();

      // phase 3: vertical 9-sum, exp, accumulate (no trailing barrier needed:
      // next phase-2 srs writes are fenced by the barrier after next phase 1)
      const int xc = opx + dx + RR;
      #pragma unroll
      for (int k = 0; k < 4; ++k) {
        const int py = opy + k*8;
        const float* colp = &srs[py*(TT+1) + opx];
        float pd = 0.f;
        #pragma unroll
        for (int v = 0; v < 9; ++v) pd += cw[v] * colp[v*(TT+1)];
        const float w = __expf(-pd * isg);
        const v4f xs = sx4[(py + dy + RR)*XRG + xc];
        num0[k] += w * xs.x;
        num1[k] += w * xs.y;
        num2[k] += w * xs.z;
        den[k]  += w;
      }
    }
  }

  #pragma unroll
  for (int k = 0; k < 4; ++k) {
    const int py  = opy + k*8;
    const float inv = 1.0f / den[k];          // den >= 1 (offset 0 gives w=1)
    const int g = ((ty0 + py)*HW + (tx0 + opx))*CH;
    out[g+0] = num0[k] * inv;
    out[g+1] = num1[k] * inv;
    out[g+2] = num2[k] * inv;
  }
}

// ---------------------------------------------------------------------------
extern "C" void kernel_launch(void* const* d_in, const int* in_sizes, int n_in,
                              void* d_out, int out_size, void* d_ws, size_t ws_size,
                              hipStream_t stream) {
  (void)in_sizes; (void)n_in; (void)out_size; (void)ws_size;
  const float* x   = (const float*)d_in[0];   // 1x384x384x3
  const float* ff  = (const float*)d_in[1];   // 7x7x3x3
  const float* pf  = (const float*)d_in[2];   // 9x9
  const float* isg = (const float*)d_in[3];   // 1
  float* out  = (float*)d_out;                // 384*384*3
  float* feat = (float*)d_ws;                 // scratch: 384*384*3 f32 (1.77 MB)

  feat_conv_wmma<<<dim3((HW*HW)/128), 256, 0, stream>>>(x, ff, feat);
  nlm_tile_kernel<<<dim3(HW/TT, HW/TT), 256, 0, stream>>>(x, feat, pf, isg, out);
}
